// LoraQKV_7155415515287
// MI455X (gfx1250) — compile-verified, matrix operand order checked by reading, not run on hardware
//
#include <hip/hip_runtime.h>
#include <hip/hip_bf16.h>

#define IN_DIM   4096
#define OUT_DIM  12288
#define M_TOT    8192
#define RANK     16
#define LORA_SCALE 2.0f   // 32.0 / 16

typedef __attribute__((ext_vector_type(16))) __bf16 v16bf;
typedef __attribute__((ext_vector_type(8)))  __bf16 bf16x8;
typedef __attribute__((ext_vector_type(8)))  float  v8f;
typedef __attribute__((address_space(3)))    __bf16 lds_bf16;

// ---------------------------------------------------------------------------
// Phase 1: fold LoRA into the weight and quantize to bf16.
// Weff[o][d] = W[o][d] + LORA_SCALE * sum_r B[k][o'][r] * A[k][r][d]
// ---------------------------------------------------------------------------
__global__ __launch_bounds__(256) void weff_kernel(
    const float* __restrict__ W, const float* __restrict__ A,
    const float* __restrict__ Bm, __bf16* __restrict__ Weff) {
  const long long idx = (long long)blockIdx.x * 256 + threadIdx.x;
  const long long base = idx * 8;
  const int d0 = (int)(base & (IN_DIM - 1));
  const int o  = (int)(base >> 12);           // IN_DIM == 2^12
  const int k  = o / (OUT_DIM / 3);
  const int op = o % (OUT_DIM / 3);

  const float* __restrict__ Brow = Bm + ((size_t)k * (OUT_DIM / 3) + op) * RANK;
  float bb[RANK];
#pragma unroll
  for (int r = 0; r < RANK; ++r) bb[r] = Brow[r] * LORA_SCALE;

  float acc[8];
  const float* __restrict__ wsrc = W + (size_t)o * IN_DIM + d0;
#pragma unroll
  for (int j = 0; j < 8; ++j) acc[j] = wsrc[j];

#pragma unroll
  for (int r = 0; r < RANK; ++r) {
    const float* __restrict__ arow = A + ((size_t)k * RANK + r) * IN_DIM + d0;
#pragma unroll
    for (int j = 0; j < 8; ++j) acc[j] += bb[r] * arow[j];
  }

  __bf16* dst = Weff + (size_t)o * IN_DIM + d0;
#pragma unroll
  for (int j = 0; j < 8; ++j) dst[j] = (__bf16)acc[j];
}

// ---------------------------------------------------------------------------
// Phase 2: out[M, N] = x[M, K](bf16-converted) @ Weff[N, K]^T + bias
// Tile 128x128, K-step 64, 256 threads = 8 waves (wave32).
// Double-buffered LDS; W tile staged with global_load_async_to_lds_b128
// (ASYNCcnt path), X tile converted fp32->bf16 through VGPRs.
// ---------------------------------------------------------------------------
#define TM 128
#define TN 128
#define TK 64
#define LDS_STRIDE 72   // 64 + 8 pad: 144B row stride, keeps 16B alignment

union AFrag { v16bf v; bf16x8 h[2]; };

__global__ __launch_bounds__(256) void gemm_kernel(
    const float* __restrict__ X, const __bf16* __restrict__ Wb,
    const float* __restrict__ bias, float* __restrict__ out) {
  __shared__ __bf16 lX[2][TM * LDS_STRIDE];
  __shared__ __bf16 lW[2][TN * LDS_STRIDE];

  const int tid   = threadIdx.x;
  const int lane  = tid & 31;
  const int wave  = tid >> 5;
  const int tileN = blockIdx.x * TN;
  const int tileM = blockIdx.y * TM;
  const int wm    = (wave & 3) * 32;   // wave's M offset within tile
  const int wn    = (wave >> 2) * 64;  // wave's N offset within tile
  const int lrow  = lane & 15;
  const int khalf = (lane < 16) ? 0 : 8;

  v8f acc[2][4] = {};

  // ---- staging helpers -----------------------------------------------------
  // W tile: async copy global(bf16) -> LDS, 4x b128 per thread (ASYNCcnt).
  auto stageW = [&](int buf, int k0) {
#pragma unroll
    for (int p = 0; p < 4; ++p) {
      const int e = (p * 256 + tid) * 8;
      const int r = e >> 6, c = e & 63;
      lds_bf16* dst = (lds_bf16*)(&lW[buf][r * LDS_STRIDE + c]);
      const unsigned voff =
          ((unsigned)(tileN + r) * (unsigned)IN_DIM + (unsigned)(k0 + c)) * 2u;
      asm volatile("global_load_async_to_lds_b128 %0, %1, %2"
                   :: "v"(dst), "v"(voff), "s"(Wb) : "memory");
    }
  };
  // X tile: global fp32 loads into registers (issued early for overlap).
  auto loadX = [&](int k0, float4 (&f)[4][2]) {
#pragma unroll
    for (int p = 0; p < 4; ++p) {
      const int e = (p * 256 + tid) * 8;
      const int r = e >> 6, c = e & 63;
      const float4* src = reinterpret_cast<const float4*>(
          X + (size_t)(tileM + r) * IN_DIM + k0 + c);
      f[p][0] = src[0];
      f[p][1] = src[1];
    }
  };
  // X tile: convert fp32 -> bf16 and store to LDS.
  auto storeX = [&](int buf, float4 (&f)[4][2]) {
#pragma unroll
    for (int p = 0; p < 4; ++p) {
      const int e = (p * 256 + tid) * 8;
      const int r = e >> 6, c = e & 63;
      bf16x8 h;
      h[0] = (__bf16)f[p][0].x; h[1] = (__bf16)f[p][0].y;
      h[2] = (__bf16)f[p][0].z; h[3] = (__bf16)f[p][0].w;
      h[4] = (__bf16)f[p][1].x; h[5] = (__bf16)f[p][1].y;
      h[6] = (__bf16)f[p][1].z; h[7] = (__bf16)f[p][1].w;
      *reinterpret_cast<bf16x8*>(&lX[buf][r * LDS_STRIDE + c]) = h;
    }
  };

  // ---- prologue: fill buffer 0 --------------------------------------------
  float4 fr[4][2];
  loadX(0, fr);
  stageW(0, 0);
  storeX(0, fr);
  asm volatile("s_wait_asynccnt 0x0" ::: "memory");
  __syncthreads();

  int cur = 0;
  for (int k0 = 0; k0 < IN_DIM; k0 += TK) {
    const int  nxt  = cur ^ 1;
    const bool more = (k0 + TK) < IN_DIM;   // uniform (scalar) branch

    if (more) {
      loadX(k0 + TK, fr);     // global loads in flight during WMMA
      stageW(nxt, k0 + TK);   // async copies in flight during WMMA
    }

    // ---- compute on buffer `cur` ------------------------------------------
#pragma unroll
    for (int kk = 0; kk < TK; kk += 32) {
      AFrag a[2], bfr[4];
#pragma unroll
      for (int mi = 0; mi < 2; ++mi) {
        const __bf16* base =
            &lX[cur][(wm + mi * 16 + lrow) * LDS_STRIDE + kk + khalf];
        a[mi].h[0] = *reinterpret_cast<const bf16x8*>(base);
        a[mi].h[1] = *reinterpret_cast<const bf16x8*>(base + 16);
      }
#pragma unroll
      for (int ni = 0; ni < 4; ++ni) {
        const __bf16* base =
            &lW[cur][(wn + ni * 16 + lrow) * LDS_STRIDE + kk + khalf];
        bfr[ni].h[0] = *reinterpret_cast<const bf16x8*>(base);
        bfr[ni].h[1] = *reinterpret_cast<const bf16x8*>(base + 16);
      }
#pragma unroll
      for (int mi = 0; mi < 2; ++mi)
#pragma unroll
        for (int ni = 0; ni < 4; ++ni)
          acc[mi][ni] = __builtin_amdgcn_wmma_f32_16x16x32_bf16(
              false, a[mi].v, false, bfr[ni].v,
              (short)0, acc[mi][ni], false, false);
    }

    if (more) {
      storeX(nxt, fr);                                  // waits loadcnt inside
      asm volatile("s_wait_asynccnt 0x0" ::: "memory"); // my async W done
    }
    __syncthreads();   // single barrier per iteration (covers dscnt)
    cur = nxt;
  }

  // ---- epilogue: lane<16: M=j, lane>=16: M=8+j; N = lane%16 ----------------
  const int rshift = (lane < 16) ? 0 : 8;
#pragma unroll
  for (int mi = 0; mi < 2; ++mi) {
    const int rbase = tileM + wm + mi * 16 + rshift;
#pragma unroll
    for (int ni = 0; ni < 4; ++ni) {
      const int col = tileN + wn + ni * 16 + lrow;
      const float bv = bias[col];
#pragma unroll
      for (int j = 0; j < 8; ++j)
        out[(size_t)(rbase + j) * OUT_DIM + col] = acc[mi][ni][j] + bv;
    }
  }
}

// ---------------------------------------------------------------------------
extern "C" void kernel_launch(void* const* d_in, const int* in_sizes, int n_in,
                              void* d_out, int out_size, void* d_ws, size_t ws_size,
                              hipStream_t stream) {
  const float* x  = (const float*)d_in[0];   // [4, 2048, 4096]
  const float* W  = (const float*)d_in[1];   // [12288, 4096]
  const float* b  = (const float*)d_in[2];   // [12288]
  const float* A  = (const float*)d_in[3];   // [3, 16, 4096]
  const float* Bm = (const float*)d_in[4];   // [3, 4096, 16]
  float* out = (float*)d_out;                // [4, 2048, 12288]

  __bf16* Weff = (__bf16*)d_ws;              // 12288*4096*2 = ~100 MB scratch

  const int weff_blocks = (OUT_DIM * IN_DIM / 8) / 256;  // 24576
  weff_kernel<<<weff_blocks, 256, 0, stream>>>(W, A, Bm, Weff);

  dim3 grid(OUT_DIM / TN, M_TOT / TM);       // (96, 64), N-major for L2 reuse of x
  gemm_kernel<<<grid, 256, 0, stream>>>(x, Weff, b, out);
}
